// RUN_GNN_55310588838560
// MI455X (gfx1250) — matrix operand order repaired and checked by reading, not compiled, not run on hardware
//
#include <hip/hip_runtime.h>

typedef __attribute__((ext_vector_type(16))) _Float16 v16h;
typedef __attribute__((ext_vector_type(8)))  _Float16 v8h;
typedef __attribute__((ext_vector_type(2)))  __fp16   h2raw;   // cvt_pkrtz result type
typedef __attribute__((ext_vector_type(8)))  float    v8f;
typedef __attribute__((ext_vector_type(4)))  unsigned int u32x4;
typedef __attribute__((ext_vector_type(8)))  int      i32x8;
typedef __attribute__((ext_vector_type(4)))  int      i32x4;

#define N_NODE 100000
#define DD 64
#define BB 512
#define UU 1000000
#define EE 2000000

// LDS / weight-pack geometry (halfs). Strides padded to 16B multiples,
// chosen so 16 lanes reading 32B rows hit mostly-distinct LDS banks.
#define RSTRIDE 264        // per-edge staging row: [h_r 0..63 | h_qr 64..127 | h_s 128..191 | rs/msg 192..255] + pad
#define GATE_KS 200        // gate_W^T row stride (K=192 + pad)
#define HT_KS   136        // ht_W^T   row stride (K=128 + pad)
#define ATT_KS  72         // Ws^T/Wqr^T/Wh^T row stride (K=64 + pad)
#define GATE_H  (128*GATE_KS)                 // 25600
#define HT_H    (64*HT_KS)                    //  8704
#define WS_H    (32*ATT_KS)                   //  2304
#define WQR_H   (32*ATT_KS)                   //  2304
#define WPACK_LDS_H (GATE_H+HT_H+WS_H+WQR_H)  // 38912 halfs staged into LDS (77824 B)
#define WH_H    (64*ATT_KS)                   //  4608 (used only by k_finish, stays in L2)
#define WPACK_TOT_H (WPACK_LDS_H+WH_H)        // 43520

// workspace byte offsets (256B aligned)
#define OFF_QREMB  0u
#define OFF_WPACK  131072u                    // BB*DD*4
#define OFF_EXP    218112u                    // + WPACK_TOT_H*2
#define OFF_UMSG   4218112u                   // + UU*4
#define OFF_AGG    260218112u                 // + UU*64*4
#define OFF_SUMEXP 285818112u                 // + N_NODE*64*4

#if defined(__has_builtin)
#if __has_builtin(__builtin_amdgcn_tensor_load_to_lds) && \
    __has_builtin(__builtin_amdgcn_s_wait_tensorcnt)
#define USE_TDM 1
#endif
#endif

__device__ __forceinline__ void lds_fence() {
  asm volatile("s_wait_dscnt 0" ::: "memory");
}

__device__ __forceinline__ v8f wmma16(v16h a, v16h b, v8f c) {
  // D = A(16x32 f16) x B(32x16 f16) + C(16x16 f32)
  return __builtin_amdgcn_wmma_f32_16x16x32_f16(false, a, false, b, (short)0, c,
                                                false, false);
}

// A-fragment (16x32 f16) from per-wave LDS staging rows.
// lane 0..15: M=lane, K = kbase+{0..7,16..23}; lane 16..31: M=lane-16, K = kbase+{8..15,24..31}
__device__ __forceinline__ v16h ldA(const _Float16* base, int kbase, int lane) {
  const int row = lane & 15;
  const int klo = (lane & 16) ? 8 : 0;
  const _Float16* p = base + row * RSTRIDE + kbase + klo;
  v8h lo = *(const v8h*)p;
  v8h hi = *(const v8h*)(p + 16);
  v16h a;
#pragma unroll
  for (int i = 0; i < 8; ++i) { a[i] = lo[i]; a[i + 8] = hi[i]; }
  return a;
}

// B-fragment (32x16 f16) from transposed weight table wT[n][k] (stride halfs).
// lane 0..15: N=nb+lane, K=kbase+{0..7,16..23}; lane 16..31: N=nb+lane-16, K=kbase+{8..15,24..31}
__device__ __forceinline__ v16h ldB(const _Float16* wT, int stride, int nb,
                                    int kbase, int lane) {
  const int n = nb + (lane & 15);
  const int kk = (lane & 16) ? 8 : 0;
  const _Float16* p = wT + n * stride + kbase + kk;
  v8h lo = *(const v8h*)p;
  v8h hi = *(const v8h*)(p + 16);
  v16h b;
#pragma unroll
  for (int i = 0; i < 8; ++i) { b[i] = lo[i]; b[i + 8] = hi[i]; }
  return b;
}

// ---------------- prep: qr_emb gather + f16 transposed weight pack ----------------
__global__ void k_prep(const float* __restrict__ rela, const int* __restrict__ q_rel,
                       const float* __restrict__ gate_W, const float* __restrict__ ht_W,
                       const float* __restrict__ Ws, const float* __restrict__ Wqr,
                       const float* __restrict__ Wh,
                       float* __restrict__ qr_emb, _Float16* __restrict__ wpack) {
  int tid = blockIdx.x * 256 + threadIdx.x;
  if (tid < BB * DD) {
    int b = tid >> 6, d = tid & 63;
    qr_emb[tid] = rela[(size_t)q_rel[b] * DD + d];
    return;
  }
  int t = tid - BB * DD;
  if (t < GATE_H) {                       // gate_W (192x128) -> [n][k]
    int n = t / GATE_KS, k = t % GATE_KS;
    wpack[t] = (_Float16)((k < 192) ? gate_W[k * 128 + n] : 0.f);
    return;
  }
  t -= GATE_H;
  if (t < HT_H) {                         // ht_W (128x64) -> [n][k]
    int n = t / HT_KS, k = t % HT_KS;
    wpack[GATE_H + t] = (_Float16)((k < 128) ? ht_W[k * 64 + n] : 0.f);
    return;
  }
  t -= HT_H;
  if (t < WS_H) {                         // Ws (64x32) -> [n][k]
    int n = t / ATT_KS, k = t % ATT_KS;
    wpack[GATE_H + HT_H + t] = (_Float16)((k < 64) ? Ws[k * 32 + n] : 0.f);
    return;
  }
  t -= WS_H;
  if (t < WQR_H) {                        // Wqr_W (64x32) -> [n][k]
    int n = t / ATT_KS, k = t % ATT_KS;
    wpack[GATE_H + HT_H + WS_H + t] = (_Float16)((k < 64) ? Wqr[k * 32 + n] : 0.f);
    return;
  }
  t -= WQR_H;
  if (t < WH_H) {                         // Wh (64x64) -> [n][k]
    int n = t / ATT_KS, k = t % ATT_KS;
    wpack[WPACK_LDS_H + t] = (_Float16)((k < 64) ? Wh[k * 64 + n] : 0.f);
  }
}

__global__ void k_zero(float4* __restrict__ p, int n4) {
  int i = blockIdx.x * 256 + threadIdx.x;
  if (i < n4) p[i] = make_float4(0.f, 0.f, 0.f, 0.f);
}

// ---------------- output #2: h_n_qr = qr_emb[node_group] ----------------
__global__ void k_hnqr(const float* __restrict__ qr_emb, const int* __restrict__ node_group,
                       float* __restrict__ out2) {
  int tid = blockIdx.x * 256 + threadIdx.x;   // N_NODE*16 threads, float4 each
  int n = tid >> 4, c4 = (tid & 15) << 2;
  float4 v = *(const float4*)(qr_emb + (size_t)node_group[n] * DD + c4);
  *(float4*)(out2 + (size_t)n * DD + c4) = v;
}

// ---------------- main fused edge kernel (16 edges per wave, WMMA) ----------------
__global__ __launch_bounds__(256) void k_edges(
    const float* __restrict__ hidden, const float* __restrict__ rela,
    const float* __restrict__ qr_emb, const _Float16* __restrict__ wpack,
    const float* __restrict__ gate_b, const float* __restrict__ ht_b,
    const float* __restrict__ wqr_b, const float* __restrict__ walpha_W,
    const float* __restrict__ walpha_b,
    const int* __restrict__ ue_ridx, const int* __restrict__ ue_rel,
    const int* __restrict__ ue_sub,
    float* __restrict__ exp_buf, float* __restrict__ u_msg) {
  extern __shared__ char smraw[];
  _Float16* wsm  = (_Float16*)smraw;                  // 38912 halfs of weights
  _Float16* stag = wsm + WPACK_LDS_H;                 // 8 waves * 16 * RSTRIDE

  const int tid = threadIdx.x;
#ifdef USE_TDM
  // Tensor Data Mover: wave 0 DMAs the whole f16 weight pack into LDS.
  // 1-D descriptor: 9728 x 8B elements, tile == tensor, lds_addr = 0.
  if (tid < 32) {
    const unsigned long long ga = (unsigned long long)wpack;
    u32x4 g0;
    g0[0] = 1u;                                        // count=1, user descriptor
    g0[1] = 0u;                                        // lds_addr = 0 (dyn-LDS base)
    g0[2] = (unsigned)(ga & 0xffffffffu);              // global_addr[31:0]
    g0[3] = (unsigned)((ga >> 32) & 0x01ffffffu) | (2u << 30);  // addr[56:32] | type=2
    i32x8 g1;
    g1[0] = 0x00030000;                                // data_size=3 (8B), no multicast
    g1[1] = (int)(9728u << 16);                        // tensor_dim0[15:0] in [31:16]
    g1[2] = 0x00010000;                                // tensor_dim1 = 1
    g1[3] = (int)(9728u << 16);                        // tile_dim0 = 9728
    g1[4] = 1;                                         // tile_dim1 = 1
    g1[5] = 9728;                                      // tensor_dim0_stride
    g1[6] = 0;
    g1[7] = 0;
    i32x4 g2 = {0, 0, 0, 0};
    i32x4 g3 = {0, 0, 0, 0};
    i32x8 g4 = {0, 0, 0, 0, 0, 0, 0, 0};
    __builtin_amdgcn_tensor_load_to_lds(g0, g1, g2, g3, g4, 0);
    __builtin_amdgcn_s_wait_tensorcnt(0);
  }
#else
  {  // cooperative fallback: 38912 halfs = 4864 x 16B
    const uint4* src = (const uint4*)wpack;
    uint4* dst = (uint4*)wsm;
    for (int i = tid; i < 4864; i += 256) dst[i] = src[i];
  }
#endif
  __syncthreads();

  const int wave = tid >> 5;
  const int lane = tid & 31;
  const int tile = blockIdx.x * 8 + wave;
  const bool active = (tile < UU / 16);               // 62500 full tiles
  const int eb = tile * 16;
  _Float16* mystag = stag + wave * 16 * RSTRIDE;

  if (active) {
    // stage 16 edges x {h_r, h_qr, h_s} as f16 rows (gathers are L2-resident);
    // branch-free source select, pk-converted 8B LDS stores
    for (int t = lane; t < 48; t += 32) {
      const int e = eb + t / 3;
      const int sel = t % 3;
      const int i0 = ue_rel[e];
      const int i1 = ue_ridx[e];
      const int i2 = ue_sub[e];
      const float* srow = (sel == 0) ? (rela + (size_t)i0 * DD)
                        : (sel == 1) ? (qr_emb + (size_t)i1 * DD)
                                     : (hidden + (size_t)i2 * DD);
      _Float16* drow = mystag + (size_t)(t / 3) * RSTRIDE + (sel << 6);
#pragma unroll
      for (int c = 0; c < 16; ++c) {
        float4 v = ((const float4*)srow)[c];
        union { h2raw h[2]; uint2 u; } pk;
        pk.h[0] = __builtin_amdgcn_cvt_pkrtz(v.x, v.y);
        pk.h[1] = __builtin_amdgcn_cvt_pkrtz(v.z, v.w);
        *(uint2*)(drow + c * 4) = pk.u;
      }
    }
  }
  __syncthreads();
  if (!active) return;

  const int colL  = lane & 15;
  const int rowHi = (lane & 16) ? 8 : 0;

  const _Float16* wGate = wsm;
  const _Float16* wHt   = wsm + GATE_H;
  const _Float16* wWs   = wsm + GATE_H + HT_H;
  const _Float16* wWqr  = wsm + GATE_H + HT_H + WS_H;

  // ---- GEMM1: g = sigmoid([h_r|h_qr|h_s] @ gate_W + gate_b)  (16x192)x(192x128)
  v8f g[8];
#pragma unroll
  for (int nt = 0; nt < 8; ++nt) {
    const float bv = gate_b[nt * 16 + colL];
#pragma unroll
    for (int i = 0; i < 8; ++i) g[nt][i] = bv;
  }
#pragma unroll
  for (int kc = 0; kc < 6; ++kc) {
    v16h a = ldA(mystag, kc * 32, lane);
#pragma unroll
    for (int nt = 0; nt < 8; ++nt)
      g[nt] = wmma16(a, ldB(wGate, GATE_KS, nt * 16, kc * 32, lane), g[nt]);
  }
#pragma unroll
  for (int nt = 0; nt < 8; ++nt)
#pragma unroll
    for (int i = 0; i < 8; ++i) g[nt][i] = 1.f / (1.f + __expf(-g[nt][i]));

  // store reset*h_s (cols 64..127 of g) into staging slot [192..256)
#pragma unroll
  for (int nt = 4; nt < 8; ++nt) {
    const int j = (nt - 4) * 16 + colL;
#pragma unroll
    for (int r = 0; r < 8; ++r) {
      const int M = r + rowHi;
      const float hs = (float)mystag[M * RSTRIDE + 128 + j];
      mystag[M * RSTRIDE + 192 + j] = (_Float16)(g[nt][r] * hs);
    }
  }
  lds_fence();

  // ---- GEMM2: cand = tanh([h_r | reset*h_s] @ ht_W + ht_b)  (16x128)x(128x64)
  v8f c2[4];
#pragma unroll
  for (int nt = 0; nt < 4; ++nt) {
    const float bv = ht_b[nt * 16 + colL];
#pragma unroll
    for (int i = 0; i < 8; ++i) c2[nt][i] = bv;
  }
  const int kb2[4] = {0, 32, 192, 224};
#pragma unroll
  for (int kc = 0; kc < 4; ++kc) {
    v16h a = ldA(mystag, kb2[kc], lane);
#pragma unroll
    for (int nt = 0; nt < 4; ++nt)
      c2[nt] = wmma16(a, ldB(wHt, HT_KS, nt * 16, kc * 32, lane), c2[nt]);
  }

  // msg = (1-update)*h_s + update*cand ; keep f32 in regs, f16 copy -> staging [192..256)
  v8f msg[4];
#pragma unroll
  for (int nt = 0; nt < 4; ++nt) {
    const int j = nt * 16 + colL;
#pragma unroll
    for (int r = 0; r < 8; ++r) {
      const int M = r + rowHi;
      const float e2 = __expf(2.f * c2[nt][r]);
      const float cand = (e2 - 1.f) / (e2 + 1.f);
      const float hs = (float)mystag[M * RSTRIDE + 128 + j];
      const float u = g[nt][r];
      const float m = (1.f - u) * hs + u * cand;
      msg[nt][r] = m;
      mystag[M * RSTRIDE + 192 + j] = (_Float16)m;
    }
  }
  lds_fence();

  // ---- GEMM3+4: attn_pre = relu(msg@Ws + h_qr@Wqr + Wqr_b)  (16x64)x(64x32) twice
  v8f c3[2];
#pragma unroll
  for (int nt = 0; nt < 2; ++nt) {
    const float bv = wqr_b[nt * 16 + colL];
#pragma unroll
    for (int i = 0; i < 8; ++i) c3[nt][i] = bv;
  }
#pragma unroll
  for (int kc = 0; kc < 2; ++kc) {
    v16h a = ldA(mystag, 192 + kc * 32, lane);          // msg
#pragma unroll
    for (int nt = 0; nt < 2; ++nt)
      c3[nt] = wmma16(a, ldB(wWs, ATT_KS, nt * 16, kc * 32, lane), c3[nt]);
  }
#pragma unroll
  for (int kc = 0; kc < 2; ++kc) {
    v16h a = ldA(mystag, 64 + kc * 32, lane);           // h_qr
#pragma unroll
    for (int nt = 0; nt < 2; ++nt)
      c3[nt] = wmma16(a, ldB(wWqr, ATT_KS, nt * 16, kc * 32, lane), c3[nt]);
  }

  // attn = relu(.)@walpha + b ; butterfly-reduce across 16-lane halves (wave32)
  const float wa0 = walpha_W[colL];
  const float wa1 = walpha_W[16 + colL];
  const float wb = walpha_b[0];
  float expa[8];
#pragma unroll
  for (int r = 0; r < 8; ++r) {
    const float t0 = fmaxf(c3[0][r], 0.f);
    const float t1 = fmaxf(c3[1][r], 0.f);
    float p = t0 * wa0 + t1 * wa1;
    p += __shfl_xor(p, 1, 32);
    p += __shfl_xor(p, 2, 32);
    p += __shfl_xor(p, 4, 32);
    p += __shfl_xor(p, 8, 32);
    expa[r] = __expf(p + wb);
  }

  // u_msg = exp_a * msg  (the only large HBM write)
#pragma unroll
  for (int nt = 0; nt < 4; ++nt)
#pragma unroll
    for (int r = 0; r < 8; ++r)
      u_msg[(size_t)(eb + r + rowHi) * DD + nt * 16 + colL] = expa[r] * msg[nt][r];

  if (colL == 0) {
#pragma unroll
    for (int r = 0; r < 8; ++r) exp_buf[eb + rowHi + r] = expa[r];
  }
}

// ---------------- segment-sum via f32 atomics into L2-resident agg ----------------
__global__ void k_scatter(const int* __restrict__ inv, const int* __restrict__ obj,
                          const float* __restrict__ exp_buf, const float* __restrict__ u_msg,
                          float* __restrict__ agg, float* __restrict__ sum_exp) {
  int tid = blockIdx.x * 256 + threadIdx.x;     // EE*16 threads
  int i = tid >> 4;
  if (i >= EE) return;
  int c4 = (tid & 15) << 2;
  int j = inv[i];
  int o = obj[i];
  float4 v = *(const float4*)(u_msg + (size_t)j * DD + c4);
  float* dst = agg + (size_t)o * DD + c4;
  atomicAdd(dst + 0, v.x);
  atomicAdd(dst + 1, v.y);
  atomicAdd(dst + 2, v.z);
  atomicAdd(dst + 3, v.w);
  if (c4 == 0) atomicAdd(sum_exp + o, exp_buf[j]);
}

// ---------------- hidden_new = relu((agg/sum_exp) @ Wh), WMMA per 16-node tile ----
__global__ __launch_bounds__(256) void k_finish(const float* __restrict__ agg,
                                                const float* __restrict__ sum_exp,
                                                const _Float16* __restrict__ whT,
                                                float* __restrict__ out) {
  const int wave = threadIdx.x >> 5;
  const int lane = threadIdx.x & 31;
  const int tile = blockIdx.x * 8 + wave;
  if (tile >= N_NODE / 16) return;
  const int nb = tile * 16;
  const int row = lane & 15;
  const int klo = (lane & 16) ? 8 : 0;
  const int colL = lane & 15;
  const int rowHi = (lane & 16) ? 8 : 0;

  const float invse = 1.f / sum_exp[nb + row];
  v16h afr[2];
#pragma unroll
  for (int kc = 0; kc < 2; ++kc) {
    const float* p = agg + (size_t)(nb + row) * DD + kc * 32 + klo;
    union { h2raw h[8]; v16h v; } af;
#pragma unroll
    for (int i = 0; i < 4; ++i) {
      af.h[i]     = __builtin_amdgcn_cvt_pkrtz(p[2 * i] * invse, p[2 * i + 1] * invse);
      af.h[4 + i] = __builtin_amdgcn_cvt_pkrtz(p[16 + 2 * i] * invse,
                                               p[16 + 2 * i + 1] * invse);
    }
    afr[kc] = af.v;
  }
  v8f c[4];
#pragma unroll
  for (int nt = 0; nt < 4; ++nt)
#pragma unroll
    for (int i = 0; i < 8; ++i) c[nt][i] = 0.f;
#pragma unroll
  for (int kc = 0; kc < 2; ++kc)
#pragma unroll
    for (int nt = 0; nt < 4; ++nt)
      c[nt] = wmma16(afr[kc], ldB(whT, ATT_KS, nt * 16, kc * 32, lane), c[nt]);
#pragma unroll
  for (int nt = 0; nt < 4; ++nt)
#pragma unroll
    for (int r = 0; r < 8; ++r)
      out[(size_t)(nb + r + rowHi) * DD + nt * 16 + colL] = fmaxf(c[nt][r], 0.f);
}

extern "C" void kernel_launch(void* const* d_in, const int* in_sizes, int n_in,
                              void* d_out, int out_size, void* d_ws, size_t ws_size,
                              hipStream_t stream) {
  (void)in_sizes; (void)n_in; (void)out_size; (void)ws_size;
  const float* hidden   = (const float*)d_in[0];
  const float* rela     = (const float*)d_in[1];
  const float* gate_W   = (const float*)d_in[2];
  const float* gate_b   = (const float*)d_in[3];
  const float* ht_W     = (const float*)d_in[4];
  const float* ht_b     = (const float*)d_in[5];
  const float* Wh       = (const float*)d_in[6];
  const float* Ws       = (const float*)d_in[7];
  const float* Wqr_W    = (const float*)d_in[8];
  const float* Wqr_b    = (const float*)d_in[9];
  const float* walpha_W = (const float*)d_in[10];
  const float* walpha_b = (const float*)d_in[11];
  const int* q_rel      = (const int*)d_in[12];
  const int* ue_ridx    = (const int*)d_in[13];
  const int* ue_rel     = (const int*)d_in[14];
  const int* ue_sub     = (const int*)d_in[15];
  const int* inv_idx    = (const int*)d_in[16];
  const int* obj        = (const int*)d_in[17];
  const int* node_group = (const int*)d_in[18];

  char* ws = (char*)d_ws;
  float*     qr_emb  = (float*)(ws + OFF_QREMB);
  _Float16*  wpack   = (_Float16*)(ws + OFF_WPACK);
  float*     exp_buf = (float*)(ws + OFF_EXP);
  float*     u_msg   = (float*)(ws + OFF_UMSG);
  float*     agg     = (float*)(ws + OFF_AGG);
  float*     sum_exp = (float*)(ws + OFF_SUMEXP);
  float* out_hidden  = (float*)d_out;
  float* out_hnqr    = out_hidden + (size_t)N_NODE * DD;

  {
    const int total = BB * DD + WPACK_TOT_H;  // 76288
    k_prep<<<(total + 255) / 256, 256, 0, stream>>>(rela, q_rel, gate_W, ht_W, Ws,
                                                    Wqr_W, Wh, qr_emb, wpack);
  }
  {
    const int n4 = (N_NODE * 65) / 4;         // zero agg + sum_exp (contiguous)
    k_zero<<<(n4 + 255) / 256, 256, 0, stream>>>((float4*)agg, n4);
  }
  k_hnqr<<<(N_NODE * 16) / 256, 256, 0, stream>>>(qr_emb, node_group, out_hnqr);
  {
    const int tiles = UU / 16;                // 62500
    const int blocks = (tiles + 7) / 8;       // 7813
    const size_t lds = (size_t)(WPACK_LDS_H + 8 * 16 * RSTRIDE) * 2;  // 145408 B
    k_edges<<<blocks, 256, lds, stream>>>(hidden, rela, qr_emb, wpack, gate_b, ht_b,
                                          Wqr_b, walpha_W, walpha_b, ue_ridx, ue_rel,
                                          ue_sub, exp_buf, u_msg);
  }
  k_scatter<<<(EE * 16) / 256, 256, 0, stream>>>(inv_idx, obj, exp_buf, u_msg, agg,
                                                 sum_exp);
  {
    const int tiles = N_NODE / 16;            // 6250
    k_finish<<<(tiles + 7) / 8, 256, 0, stream>>>(agg, sum_exp, wpack + WPACK_LDS_H,
                                                  out_hidden);
  }
}
